// SelfContextCluster_9405978378726
// MI455X (gfx1250) — matrix operand order repaired and checked by reading, not compiled, not run on hardware
//
#include <hip/hip_runtime.h>

// ---------------------------------------------------------------------------
// MI455X (gfx1250) implementation.
//   K1: xp = x @ W_proj^T + b_proj   -> bf16, layout (m=n*8+g, p, c) c in [0,64)
//   K2: per-m (128 WGs): maxpool->anchors, l2norm, sim via WMMA bf16,
//       sigmoid+argmax one-hot (shfl), LDS-atomic anchor aggregation,
//       normalize, msg gather -> bf16 (row=pixel, col=g*32+d)
//   K3: out = msg @ W_out^T + b_out  -> f32 d_out
// All intermediates bf16 and L2-resident (xp 67MB, msg 33.5MB).
// ---------------------------------------------------------------------------

typedef __bf16 bf16_t;
typedef __attribute__((ext_vector_type(16))) bf16_t v16bf;
typedef __attribute__((ext_vector_type(8)))  bf16_t v8bf;
typedef __attribute__((ext_vector_type(8)))  float  v8f;
typedef __attribute__((ext_vector_type(4)))  float  v4f;

__device__ __forceinline__ v8f wmma_bf16(v16bf a, v16bf b, v8f c) {
  // D = A(16x32) * B(32x16) + C(16x16), f32 accumulate
  return __builtin_amdgcn_wmma_f32_16x16x32_bf16(
      /*neg_a=*/false, a, /*neg_b=*/false, b,
      /*c_mod=*/(short)0, c, /*reuse_a=*/false, /*reuse_b=*/false);
}

// A-fragment (16x32 bf16): lane r=lane&15 is row M; in this code's k order the lane
// holds K = ko..ko+15 contiguously, ko = (lane>>4)*16.
__device__ __forceinline__ v16bf load_a_frag(const bf16_t* base, int stride) {
  const int lane = threadIdx.x & 31;
  const int r    = lane & 15;
  const int ko   = (lane >> 4) << 4;
  const bf16_t* p = base + (size_t)r * stride;
  v8bf lo = *(const v8bf*)(p + ko);
  v8bf hi = *(const v8bf*)(p + 8 + ko);
  v16bf f;
#pragma unroll
  for (int i = 0; i < 8; ++i) { f[i] = lo[i]; f[i + 8] = hi[i]; }
  return f;
}

// B-fragment (32x16 bf16) from row-major (col, k) storage: lane col=lane&15,
// holds K = (lane>>4)*16 .. +15 contiguously (same k order as load_a_frag).
__device__ __forceinline__ v16bf load_b_frag(const bf16_t* base, int stride) {
  const int lane = threadIdx.x & 31;
  const int col  = lane & 15;
  const int ko   = (lane >> 4) << 4;
  return *(const v16bf*)(base + (size_t)col * stride + ko);
}

// ---------------------------------------------------------------------------
// K1: GEMM1  (65536 x 512, K=256), output scattered to (m,p,c) bf16 layout
// ---------------------------------------------------------------------------
#define BM 128
#define BN 64
#define BK 32
#define LDA (BK + 8)

__global__ __launch_bounds__(256) void gemm1_proj(
    const float* __restrict__ x,    // 65536 x 256
    const float* __restrict__ Wp,   // 512 x 256 (row-major -> B is (col,k))
    const float* __restrict__ bp,   // 512
    bf16_t* __restrict__ xp)        // ((n*8+g)*4096 + p)*64 + c
{
  __shared__ bf16_t As[BM][LDA];
  __shared__ bf16_t Bs[BN][LDA];

  const int tid  = threadIdx.x;
  const int lane = tid & 31;
  const int wv   = tid >> 5;   // 8 waves: 4 (M) x 2 (N)
  const int wm   = wv & 3;
  const int wn   = wv >> 2;
  const int rb   = blockIdx.y * BM;
  const int cb   = blockIdx.x * BN;

  v8f acc[2][2];
#pragma unroll
  for (int i = 0; i < 2; ++i)
#pragma unroll
    for (int j = 0; j < 2; ++j) acc[i][j] = (v8f){};

  const int arow = tid >> 1, aseg = (tid & 1) * 16;
  const int brow = tid >> 2, bseg = (tid & 3) * 8;

  for (int kb = 0; kb < 256; kb += BK) {
    { // stage A tile (f32 -> bf16)
      const float* src = x + (size_t)(rb + arow) * 256 + kb + aseg;
      if (kb + BK < 256) __builtin_prefetch(src + BK, 0, 1);
      const v4f* s4 = (const v4f*)src;
      v4f f0 = s4[0], f1 = s4[1], f2 = s4[2], f3 = s4[3];
      v8bf o0, o1;
#pragma unroll
      for (int j = 0; j < 4; ++j) {
        o0[j] = (bf16_t)f0[j]; o0[4 + j] = (bf16_t)f1[j];
        o1[j] = (bf16_t)f2[j]; o1[4 + j] = (bf16_t)f3[j];
      }
      *(v8bf*)&As[arow][aseg]     = o0;
      *(v8bf*)&As[arow][aseg + 8] = o1;
    }
    { // stage B tile (weights, f32 -> bf16), stored (col, k)
      const float* src = Wp + (size_t)(cb + brow) * 256 + kb + bseg;
      const v4f* s4 = (const v4f*)src;
      v4f f0 = s4[0], f1 = s4[1];
      v8bf o;
#pragma unroll
      for (int j = 0; j < 4; ++j) { o[j] = (bf16_t)f0[j]; o[4 + j] = (bf16_t)f1[j]; }
      *(v8bf*)&Bs[brow][bseg] = o;
    }
    __syncthreads();
    v16bf a0 = load_a_frag(&As[wm * 32][0],      LDA);
    v16bf a1 = load_a_frag(&As[wm * 32 + 16][0], LDA);
    v16bf b0 = load_b_frag(&Bs[wn * 32][0],      LDA);
    v16bf b1 = load_b_frag(&Bs[wn * 32 + 16][0], LDA);
    acc[0][0] = wmma_bf16(a0, b0, acc[0][0]);
    acc[0][1] = wmma_bf16(a0, b1, acc[0][1]);
    acc[1][0] = wmma_bf16(a1, b0, acc[1][0]);
    acc[1][1] = wmma_bf16(a1, b1, acc[1][1]);
    __syncthreads();
  }

  // epilogue: +bias, scatter to (m, p, c) bf16 layout
#pragma unroll
  for (int fm = 0; fm < 2; ++fm)
#pragma unroll
    for (int fn = 0; fn < 2; ++fn) {
      const int colg = cb + wn * 32 + fn * 16 + (lane & 15);
      const float bias = bp[colg];
      const int g = colg >> 6, c = colg & 63;
#pragma unroll
      for (int v = 0; v < 8; ++v) {
        const int rowg = rb + wm * 32 + fm * 16 + v + 8 * (lane >> 4);
        const int n = rowg >> 12, p = rowg & 4095;
        const size_t idx = (((size_t)((n * 8 + g) * 4096 + p)) << 6) + c;
        xp[idx] = (bf16_t)(acc[fm][fn][v] + bias);
      }
    }
}

// ---------------------------------------------------------------------------
// K2: per-m clustering (one workgroup of 512 threads per m, 128 WGs)
// ---------------------------------------------------------------------------
__global__ __launch_bounds__(512) void cluster_k(
    const bf16_t* __restrict__ xp,   // (m, p=4096, c=64)
    const float* __restrict__ alpha,
    const float* __restrict__ beta,
    bf16_t* __restrict__ msg)        // (row = n*4096+p, col = g*32+d), 256 cols
{
  __shared__ float         pooled_s[64][64];  // anchors x channels
  __shared__ bf16_t        p1n[64][32];       // normalized anchor points (B frags)
  __shared__ float         val1[64][32];
  __shared__ float         agg_s[64][32];
  __shared__ float         ssum[64];
  __shared__ float         s_arr[4096];
  __shared__ unsigned char idx_arr[4096];

  const int m    = blockIdx.x;
  const int tid  = threadIdx.x;
  const int lane = tid & 31;
  const int wv   = tid >> 5;  // 16 waves
  const bf16_t* xm = xp + (size_t)m * 4096 * 64;

  // ---- 8x8 max-pool into 64 anchors (8 threads per anchor, 8 ch each) ----
  {
    const int a  = tid >> 3;
    const int c0 = (tid & 7) * 8;
    const int ai = a >> 3, aj = a & 7;
    float mx[8];
#pragma unroll
    for (int i = 0; i < 8; ++i) mx[i] = -3.4e38f;
    for (int hr = 0; hr < 8; ++hr) {
      const int prow = (ai * 8 + hr) * 64 + aj * 8;
#pragma unroll
      for (int wc = 0; wc < 8; ++wc) {
        v8bf v = *(const v8bf*)(xm + (size_t)(prow + wc) * 64 + c0);
#pragma unroll
        for (int i = 0; i < 8; ++i) mx[i] = fmaxf(mx[i], (float)v[i]);
      }
    }
#pragma unroll
    for (int i = 0; i < 8; ++i) pooled_s[a][c0 + i] = mx[i];
  }
  for (int i = tid; i < 2048; i += 512) agg_s[i >> 5][i & 31] = 0.0f;
  if (tid < 64) ssum[tid] = 0.0f;
  __syncthreads();

  // ---- L2-normalize point1, keep value1 ----
  if (tid < 64) {
    const int a = tid;
    float ss = 0.0f;
    for (int d = 0; d < 32; ++d) { float t = pooled_s[a][d]; ss += t * t; }
    const float rn = 1.0f / fmaxf(sqrtf(ss), 1e-12f);
    for (int d = 0; d < 32; ++d) {
      p1n[a][d]  = (bf16_t)(pooled_s[a][d] * rn);
      val1[a][d] = pooled_s[a][32 + d];
    }
  }
  __syncthreads();

  const float a0 = alpha[0], b0 = beta[0];

  // B fragments: 4 anchor tiles of 16, kept in registers for all pixel tiles
  v16bf bfr[4];
#pragma unroll
  for (int t = 0; t < 4; ++t)
    bfr[t] = *(const v16bf*)(&p1n[t * 16 + (lane & 15)][(lane >> 4) * 16]);

  // ---- pass 1: sim (WMMA) + sigmoid + argmax one-hot + anchor aggregation ----
  for (int it = 0; it < 16; ++it) {
    const int T = wv * 16 + it;        // 256 tiles of 16 pixels
    const int pbase = T * 16;

    // A fragment: l2-normalized point0 for 16 pixels (K = 32 channels)
    const int r  = lane & 15;
    const int ko = (lane >> 4) * 16;
    const bf16_t* px = xm + (size_t)(pbase + r) * 64;
    v8bf lo = *(const v8bf*)(px + ko);
    v8bf hi = *(const v8bf*)(px + 8 + ko);
    float lf[8], hf[8], ss = 0.0f;
#pragma unroll
    for (int j = 0; j < 8; ++j) {
      lf[j] = (float)lo[j]; hf[j] = (float)hi[j];
      ss += lf[j] * lf[j] + hf[j] * hf[j];
    }
    ss += __shfl_xor(ss, 16, 32);      // partner lane holds the other 16 ch
    const float rn = 1.0f / fmaxf(sqrtf(ss), 1e-12f);
    v16bf af;
#pragma unroll
    for (int j = 0; j < 8; ++j) {
      af[j]     = (bf16_t)(lf[j] * rn);
      af[j + 8] = (bf16_t)(hf[j] * rn);
    }

    v8f acc[4];
#pragma unroll
    for (int t = 0; t < 4; ++t) { v8f z = (v8f){}; acc[t] = wmma_bf16(af, bfr[t], z); }

    // sigmoid(alpha*sim + beta)
    float sg[4][8];
#pragma unroll
    for (int t = 0; t < 4; ++t)
#pragma unroll
      for (int v = 0; v < 8; ++v)
        sg[t][v] = 1.0f / (1.0f + expf(-(a0 * acc[t][v] + b0)));

    // argmax over 64 anchors per pixel (16 lanes per pixel half)
    float sP[8]; int iP[8];
#pragma unroll
    for (int v = 0; v < 8; ++v) {
      float bv = -1.0f; int bi = 0;
#pragma unroll
      for (int t = 0; t < 4; ++t) {
        const float xv = sg[t][v];
        const int aa = t * 16 + (lane & 15);
        if (xv > bv) { bv = xv; bi = aa; }
      }
#pragma unroll
      for (int off = 1; off < 16; off <<= 1) {
        const float ov = __shfl_xor(bv, off, 32);
        const int   oi = __shfl_xor(bi, off, 32);
        if (ov > bv || (ov == bv && oi < bi)) { bv = ov; bi = oi; }
      }
      if ((lane & 15) == 0) {
        const int pr = pbase + v + 8 * (lane >> 4);
        s_arr[pr]   = bv;
        idx_arr[pr] = (unsigned char)bi;
      }
      sP[v] = bv; iP[v] = bi;
    }

    // anchor aggregation: lane == channel d; LDS float atomics (ds_add_f32)
#pragma unroll
    for (int rr = 0; rr < 16; ++rr) {
      const int src = (rr >> 3) * 16;
      const float s = __shfl(sP[rr & 7], src, 32);
      const int  id = __shfl(iP[rr & 7], src, 32);
      const float v0 = (float)xm[(size_t)(pbase + rr) * 64 + 32 + lane];
      atomicAdd(&agg_s[id][lane], v0 * s);
      if (lane == 0) atomicAdd(&ssum[id], s);
    }
  }
  __syncthreads();

  // ---- finalize: agg = (agg + value1) / (ssum + 1) ----
  for (int i = tid; i < 2048; i += 512) {
    const int aa = i >> 5, d = i & 31;
    agg_s[aa][d] = (agg_s[aa][d] + val1[aa][d]) / (ssum[aa] + 1.0f);
  }
  __syncthreads();

  // ---- pass 2: msg[p, d] = s_p * agg[idx_p, d], write bf16 GEMM2 operand ----
  const int n = m >> 3, g = m & 7;
  for (int p = tid; p < 4096; p += 512) {
    const float s = s_arr[p];
    const int  id = idx_arr[p];
    bf16_t* dst = msg + ((size_t)(n * 4096 + p) * 256) + g * 32;
#pragma unroll
    for (int ch = 0; ch < 4; ++ch) {
      v8bf o;
#pragma unroll
      for (int j = 0; j < 8; ++j) o[j] = (bf16_t)(s * agg_s[id][ch * 8 + j]);
      *(v8bf*)(dst + ch * 8) = o;
    }
  }
}

// ---------------------------------------------------------------------------
// K3: GEMM2  (65536 x 256, K=256), bf16 msg * bf16 W_out -> f32 d_out
// ---------------------------------------------------------------------------
__global__ __launch_bounds__(256) void gemm2_out(
    const bf16_t* __restrict__ msg,  // 65536 x 256 bf16
    const float* __restrict__ Wo,    // 256 x 256 f32
    const float* __restrict__ bo,    // 256
    float* __restrict__ out)         // 65536 x 256 f32
{
  __shared__ bf16_t As[BM][LDA];
  __shared__ bf16_t Bs[BN][LDA];

  const int tid  = threadIdx.x;
  const int lane = tid & 31;
  const int wv   = tid >> 5;
  const int wm   = wv & 3;
  const int wn   = wv >> 2;
  const int rb   = blockIdx.y * BM;
  const int cb   = blockIdx.x * BN;

  v8f acc[2][2];
#pragma unroll
  for (int i = 0; i < 2; ++i)
#pragma unroll
    for (int j = 0; j < 2; ++j) acc[i][j] = (v8f){};

  const int arow = tid >> 1, aseg = (tid & 1) * 16;
  const int brow = tid >> 2, bseg = (tid & 3) * 8;

  for (int kb = 0; kb < 256; kb += BK) {
    { // stage A tile (already bf16)
      const bf16_t* src = msg + (size_t)(rb + arow) * 256 + kb + aseg;
      if (kb + BK < 256) __builtin_prefetch(src + BK, 0, 1);
      v8bf o0 = *(const v8bf*)(src);
      v8bf o1 = *(const v8bf*)(src + 8);
      *(v8bf*)&As[arow][aseg]     = o0;
      *(v8bf*)&As[arow][aseg + 8] = o1;
    }
    { // stage B tile (f32 -> bf16), (col, k)
      const float* src = Wo + (size_t)(cb + brow) * 256 + kb + bseg;
      const v4f* s4 = (const v4f*)src;
      v4f f0 = s4[0], f1 = s4[1];
      v8bf o;
#pragma unroll
      for (int j = 0; j < 4; ++j) { o[j] = (bf16_t)f0[j]; o[4 + j] = (bf16_t)f1[j]; }
      *(v8bf*)&Bs[brow][bseg] = o;
    }
    __syncthreads();
    v16bf a0 = load_a_frag(&As[wm * 32][0],      LDA);
    v16bf a1 = load_a_frag(&As[wm * 32 + 16][0], LDA);
    v16bf b0 = load_b_frag(&Bs[wn * 32][0],      LDA);
    v16bf b1 = load_b_frag(&Bs[wn * 32 + 16][0], LDA);
    acc[0][0] = wmma_bf16(a0, b0, acc[0][0]);
    acc[0][1] = wmma_bf16(a0, b1, acc[0][1]);
    acc[1][0] = wmma_bf16(a1, b0, acc[1][0]);
    acc[1][1] = wmma_bf16(a1, b1, acc[1][1]);
    __syncthreads();
  }

#pragma unroll
  for (int fm = 0; fm < 2; ++fm)
#pragma unroll
    for (int fn = 0; fn < 2; ++fn) {
      const int colg = cb + wn * 32 + fn * 16 + (lane & 15);
      const float bias = bo[colg];
#pragma unroll
      for (int v = 0; v < 8; ++v) {
        const int rowg = rb + wm * 32 + fm * 16 + v + 8 * (lane >> 4);
        out[(size_t)rowg * 256 + colg] = acc[fm][fn][v] + bias;
      }
    }
}

// ---------------------------------------------------------------------------
extern "C" void kernel_launch(void* const* d_in, const int* in_sizes, int n_in,
                              void* d_out, int out_size, void* d_ws, size_t ws_size,
                              hipStream_t stream) {
  (void)in_sizes; (void)n_in; (void)out_size; (void)ws_size;
  const float* x     = (const float*)d_in[0];  // (16,64,64,256)
  const float* Wp    = (const float*)d_in[1];  // (512,256)
  const float* bp    = (const float*)d_in[2];  // (512)
  const float* Wo    = (const float*)d_in[3];  // (256,256)
  const float* bo    = (const float*)d_in[4];  // (256)
  const float* alpha = (const float*)d_in[5];
  const float* beta  = (const float*)d_in[6];
  float* out = (float*)d_out;

  bf16_t* xp  = (bf16_t*)d_ws;                                   // 128*4096*64 bf16 = 64 MiB
  bf16_t* msg = (bf16_t*)((char*)d_ws + (size_t)128 * 4096 * 64 * sizeof(bf16_t)); // 32 MiB

  gemm1_proj<<<dim3(512 / BN, 65536 / BM), 256, 0, stream>>>(x, Wp, bp, xp);
  cluster_k<<<dim3(128), 512, 0, stream>>>(xp, alpha, beta, msg);
  gemm2_out<<<dim3(256 / BN, 65536 / BM), 256, 0, stream>>>(msg, Wo, bo, out);
}
